// MultiHeadedSelfAttention_60292750901789
// MI455X (gfx1250) — matrix operand order repaired and checked by reading, not compiled, not run on hardware
//
#include <hip/hip_runtime.h>
#include <math.h>

// Problem constants (from reference): B=2, S=2048, D=1024, H=8, DH=128
#define Bsz 2
#define Ssz 2048
#define Dsz 1024
#define Hsz 8
#define DHsz 128
#define NEGV (-4294967295.0f)   // -2^32+1 from reference (rounds to -2^32 in f32)

typedef __attribute__((ext_vector_type(16))) __bf16        v16bf;
typedef __attribute__((ext_vector_type(8)))  float         v8f;
typedef __attribute__((ext_vector_type(8)))  unsigned int  v8u;

// Hardware BF16 conversion (v_cvt_pk_bf16_f32 on gfx1250)
__device__ __forceinline__ unsigned short f2bf(float f) {
    return __builtin_bit_cast(unsigned short, (__bf16)f);
}
__device__ __forceinline__ unsigned int f2bf2(float lo, float hi) {
    unsigned int a = f2bf(lo), b = f2bf(hi);
    return a | (b << 16);
}
__device__ __forceinline__ v16bf wmma_bf16_frag(const v8u& u) {
    return __builtin_bit_cast(v16bf, u);
}

// ---------------------------------------------------------------------------
// Kernel 1: QKV projection.  x[4096,1024] (f32) x W[1024,1024] (f32) + bias.
// Output: Q,K as bf16 [B,H,S,DH]; V transposed as bf16 [B,H,DH,S].
// Workgroup: 256 threads (8 waves), tile 128(M) x 128(N), K-step 32.
// ---------------------------------------------------------------------------
__global__ __launch_bounds__(256) void qkv_proj_kernel(
    const float* __restrict__ x,
    const float* __restrict__ Wq, const float* __restrict__ Wk, const float* __restrict__ Wv,
    const float* __restrict__ bq, const float* __restrict__ bk, const float* __restrict__ bv,
    unsigned short* __restrict__ Qb, unsigned short* __restrict__ Kb,
    unsigned short* __restrict__ Vt)
{
    constexpr int LDP = 36;                  // padded LDS stride (dword aligned, conflict-free)
    __shared__ unsigned short sA[128 * LDP]; // A tile 128 x 32 (bf16)
    __shared__ unsigned short sB[128 * LDP]; // B tile stored transposed: [n][k] 128 x 32

    const int t   = threadIdx.x;
    const int w   = t >> 5;                  // wave id 0..7
    const int l   = t & 31;                  // lane
    const int l16 = l & 15;
    const int kh  = l >> 4;

    const int mbase = blockIdx.x * 128;
    const int nbase = blockIdx.y * 128;
    const int proj  = blockIdx.z;            // 0=Q 1=K 2=V

    const float* W  = (proj == 0) ? Wq : (proj == 1) ? Wk : Wv;
    const float* bi = (proj == 0) ? bq : (proj == 1) ? bk : bv;

    v8f acc[8];
#pragma unroll
    for (int nt = 0; nt < 8; ++nt) acc[nt] = (v8f)0.0f;

    for (int k0 = 0; k0 < Dsz; k0 += 32) {
        // --- prefetch next K-tile (global_prefetch_b8) ---
        if (k0 + 32 < Dsz) {
            __builtin_prefetch(&x[(size_t)(mbase + (t >> 1)) * Dsz + k0 + 32 + (t & 1) * 16], 0, 1);
            __builtin_prefetch(&W[(size_t)(k0 + 32 + (t >> 4)) * Dsz + nbase + (t & 15) * 8], 0, 1);
        }

        // --- stage A: x[mbase..+127][k0..+31] -> bf16 pairs in LDS ---
#pragma unroll
        for (int i = 0; i < 8; ++i) {
            int idx = i * 256 + t;                 // 2048 f32 pairs
            int r = idx >> 4, c = (idx & 15) * 2;
            const float* xp = &x[(size_t)(mbase + r) * Dsz + k0 + c];
            *(unsigned int*)&sA[r * LDP + c] = f2bf2(xp[0], xp[1]);
        }
        // --- stage B transposed: W[k0..+31][nbase..+127] -> sB[n][k] pairs ---
#pragma unroll
        for (int i = 0; i < 8; ++i) {
            int idx = i * 256 + t;                 // 16 k-pairs x 128 n
            int n = idx & 127, k = (idx >> 7) * 2;
            const float* wp = &W[(size_t)(k0 + k) * Dsz + nbase + n];
            *(unsigned int*)&sB[n * LDP + k] = f2bf2(wp[0], wp[Dsz]);
        }
        __syncthreads();

        // --- A fragment (16x32 bf16, contiguous k = 16*kh + s order) for this wave's 16-row strip ---
        const int mrow = w * 16 + l16;
        v8u au;
#pragma unroll
        for (int j = 0; j < 8; ++j) {
            int kk = 2 * j + 16 * kh;
            au[j] = *(const unsigned int*)&sA[mrow * LDP + kk];
        }
        v16bf afrag = wmma_bf16_frag(au);

#pragma unroll
        for (int nt = 0; nt < 8; ++nt) {
            int n = nt * 16 + l16;
            v8u bu;
#pragma unroll
            for (int j = 0; j < 8; ++j) {
                int kk = 2 * j + 16 * kh;
                bu[j] = *(const unsigned int*)&sB[n * LDP + kk];
            }
            v16bf bfrag = wmma_bf16_frag(bu);
            acc[nt] = __builtin_amdgcn_wmma_f32_16x16x32_bf16(
                false, afrag, false, bfrag, (short)0, acc[nt], false, false);
        }
        __syncthreads();
    }

    // --- epilogue: add bias, convert to bf16, scatter-store ---
#pragma unroll
    for (int nt = 0; nt < 8; ++nt) {
        int N = nbase + nt * 16 + l16;
        int h = N >> 7, d = N & (DHsz - 1);
        float bias = bi[N];
#pragma unroll
        for (int r = 0; r < 8; ++r) {
            int M = mbase + w * 16 + r + 8 * kh;
            int b = M >> 11, s = M & (Ssz - 1);
            unsigned short hv = f2bf(acc[nt][r] + bias);
            if (proj == 0)
                Qb[((size_t)(b * Hsz + h) * Ssz + s) * DHsz + d] = hv;
            else if (proj == 1)
                Kb[((size_t)(b * Hsz + h) * Ssz + s) * DHsz + d] = hv;
            else
                Vt[((size_t)(b * Hsz + h) * DHsz + d) * Ssz + s] = hv;
        }
    }
}

// ---------------------------------------------------------------------------
// Kernel 2: causal flash attention.  One wave per (b, h, 16-query tile).
// Online softmax over 32-key chunks; 16 bf16 WMMAs per chunk.
// ---------------------------------------------------------------------------
__global__ __launch_bounds__(32) void attn_kernel(
    const unsigned short* __restrict__ Qb,
    const unsigned short* __restrict__ Kb,
    const unsigned short* __restrict__ Vt,
    const unsigned char*  __restrict__ mask,
    float* __restrict__ out)
{
    constexpr int LDP = 36;
    __shared__ unsigned short sP[16 * LDP];  // P tile 16x32 bf16 (padded)

    const int l   = threadIdx.x;
    const int l16 = l & 15;
    const int kh  = l >> 4;

    const int qt = blockIdx.x & (Ssz / 16 - 1);           // 0..127
    const int h  = (blockIdx.x >> 7) & (Hsz - 1);
    const int b  = blockIdx.x >> 10;
    const int qbase = qt * 16;

    const unsigned short* Qp = Qb + ((size_t)(b * Hsz + h) * Ssz + qbase) * DHsz;
    const unsigned short* Kp = Kb + (size_t)(b * Hsz + h) * Ssz * DHsz;
    const unsigned short* Vp = Vt + (size_t)(b * Hsz + h) * DHsz * Ssz;

    // --- preload Q fragments (4 d-chunks of 32) straight from global ---
    v16bf qa[4];
#pragma unroll
    for (int c = 0; c < 4; ++c) {
        v8u u;
#pragma unroll
        for (int j = 0; j < 8; ++j) {
            int kk = 32 * c + 2 * j + 16 * kh;
            u[j] = *(const unsigned int*)(Qp + (size_t)l16 * DHsz + kk);
        }
        qa[c] = wmma_bf16_frag(u);
    }

    float mrow[8], lsum[8];
#pragma unroll
    for (int r = 0; r < 8; ++r) { mrow[r] = -INFINITY; lsum[r] = 0.0f; }
    v8f o[8];
#pragma unroll
    for (int nt = 0; nt < 8; ++nt) o[nt] = (v8f)0.0f;

    const float scale = 0.08838834764831845f;   // 1/sqrt(128)
    const int nch = (qbase + 47) >> 5;          // key chunks covering causal span

    for (int ch = 0; ch < nch; ++ch) {
        const int kvb = ch * 32;

        // --- scores: two 16x16 tiles, contraction over DH=128 ---
        v8f s0 = (v8f)0.0f, s1 = (v8f)0.0f;
#pragma unroll
        for (int t2 = 0; t2 < 2; ++t2) {
            const int colb = kvb + 16 * t2;
            const unsigned short* Krow = Kp + (size_t)(colb + l16) * DHsz;
            v8f cc = (v8f)0.0f;
#pragma unroll
            for (int c = 0; c < 4; ++c) {
                v8u u;
#pragma unroll
                for (int j = 0; j < 8; ++j)
                    u[j] = *(const unsigned int*)(Krow + 32 * c + 2 * j + 16 * kh);
                v16bf kf = wmma_bf16_frag(u);
                cc = __builtin_amdgcn_wmma_f32_16x16x32_bf16(
                    false, qa[c], false, kf, (short)0, cc, false, false);
            }
            if (t2 == 0) s0 = cc; else s1 = cc;
        }

        // --- mask + online softmax (rows r+8*kh live in this half-wave) ---
        const int kk0 = kvb + l16, kk1 = kvb + 16 + l16;
        const bool km0 = mask[b * Ssz + kk0] != 0;
        const bool km1 = mask[b * Ssz + kk1] != 0;
        float alph[8];
#pragma unroll
        for (int r = 0; r < 8; ++r) {
            const int qg = qbase + r + 8 * kh;
            float a0 = s0[r] * scale; if (kk0 > qg || !km0) a0 = NEGV;
            float a1 = s1[r] * scale; if (kk1 > qg || !km1) a1 = NEGV;

            float mx = fmaxf(a0, a1);
#pragma unroll
            for (int off = 1; off < 16; off <<= 1)
                mx = fmaxf(mx, __shfl_xor(mx, off, 16));
            const float mn = fmaxf(mrow[r], mx);
            const float alpha = __expf(mrow[r] - mn);
            mrow[r] = mn;

            const float p0 = __expf(a0 - mn);
            const float p1 = __expf(a1 - mn);
            float rs = p0 + p1;
#pragma unroll
            for (int off = 1; off < 16; off <<= 1)
                rs += __shfl_xor(rs, off, 16);
            lsum[r] = lsum[r] * alpha + rs;
            alph[r] = alpha;

            sP[(r + 8 * kh) * LDP + l16]      = f2bf(p0);
            sP[(r + 8 * kh) * LDP + 16 + l16] = f2bf(p1);
        }
#pragma unroll
        for (int nt = 0; nt < 8; ++nt)
#pragma unroll
            for (int r = 0; r < 8; ++r) o[nt][r] *= alph[r];

        __syncthreads();   // single wave: enforces LDS store->load ordering

        // --- P fragment (A layout 16x32) from LDS ---
        v8u pu;
#pragma unroll
        for (int j = 0; j < 8; ++j) {
            int kk = 2 * j + 16 * kh;
            pu[j] = *(const unsigned int*)&sP[l16 * LDP + kk];
        }
        v16bf pa = wmma_bf16_frag(pu);
        __syncthreads();   // protect sP before next chunk's writes

        // --- O += P (16x32) x V-chunk (32x128), 8 WMMAs from V^T rows ---
#pragma unroll
        for (int nt = 0; nt < 8; ++nt) {
            const unsigned short* Vrow = Vp + (size_t)(nt * 16 + l16) * Ssz + kvb;
            v8u u;
#pragma unroll
            for (int j = 0; j < 8; ++j)
                u[j] = *(const unsigned int*)(Vrow + 2 * j + 16 * kh);
            v16bf vf = wmma_bf16_frag(u);
            o[nt] = __builtin_amdgcn_wmma_f32_16x16x32_bf16(
                false, pa, false, vf, (short)0, o[nt], false, false);
        }
    }

    // --- epilogue: normalize, query mask, store f32 [B,S,D] ---
    float rinv[8], qmf[8];
#pragma unroll
    for (int r = 0; r < 8; ++r) {
        const int qg = qbase + r + 8 * kh;
        rinv[r] = 1.0f / lsum[r];
        qmf[r]  = mask[b * Ssz + qg] ? 1.0f : 0.0f;
    }
#pragma unroll
    for (int nt = 0; nt < 8; ++nt) {
        const int d = nt * 16 + l16;
#pragma unroll
        for (int r = 0; r < 8; ++r) {
            const int qg = qbase + r + 8 * kh;
            out[((size_t)b * Ssz + qg) * Dsz + h * DHsz + d] = o[nt][r] * rinv[r] * qmf[r];
        }
    }
}

// ---------------------------------------------------------------------------
extern "C" void kernel_launch(void* const* d_in, const int* in_sizes, int n_in,
                              void* d_out, int out_size, void* d_ws, size_t ws_size,
                              hipStream_t stream) {
    const float*         x    = (const float*)d_in[0];
    const unsigned char* mask = (const unsigned char*)d_in[1];
    const float*         Wq   = (const float*)d_in[2];
    const float*         bq   = (const float*)d_in[3];
    const float*         Wk   = (const float*)d_in[4];
    const float*         bk   = (const float*)d_in[5];
    const float*         Wv   = (const float*)d_in[6];
    const float*         bv   = (const float*)d_in[7];
    float*               out  = (float*)d_out;

    const size_t elems = (size_t)Bsz * Hsz * Ssz * DHsz;   // 4,194,304
    unsigned short* Qb = (unsigned short*)d_ws;
    unsigned short* Kb = Qb + elems;
    unsigned short* Vt = Kb + elems;

    dim3 gP(Bsz * Ssz / 128, Dsz / 128, 3);                // 32 x 8 x 3
    qkv_proj_kernel<<<gP, 256, 0, stream>>>(x, Wq, Wk, Wv, bq, bk, bv, Qb, Kb, Vt);

    dim3 gA(Bsz * Hsz * (Ssz / 16));                       // 2048 one-wave blocks
    attn_kernel<<<gA, 32, 0, stream>>>(Qb, Kb, Vt, mask, out);
}
